// MultiheadAttention_46505905881309
// MI455X (gfx1250) — compile-verified, matrix operand order checked by reading, not run on hardware
//
#include <hip/hip_runtime.h>
#include <hip/hip_bf16.h>

typedef __attribute__((ext_vector_type(16))) __bf16 v16bf;
typedef __attribute__((ext_vector_type(8)))  __bf16 v8bf;
typedef __attribute__((ext_vector_type(8)))  float  v8f;

#define T_LEN 2048
#define BATCH 2
#define EMB   1024
#define HEADS 16
#define HDIM  64

// ---------- helpers ----------

__device__ __forceinline__ __bf16 to_bf16(float f) {
  // round-to-nearest-even fp32 -> bf16
  unsigned int u = __builtin_bit_cast(unsigned int, f);
  unsigned int r = u + 0x7FFFu + ((u >> 16) & 1u);
  unsigned short h = (unsigned short)(r >> 16);
  return __builtin_bit_cast(__bf16, h);
}

// Load a 16x32 bf16 WMMA fragment (A-operand layout; identical lane mapping is
// used for the B-operand when the "row" index is the N column and ld strides K).
// Lane l<16: row = base_row + l,     K-chunk {0..7, 16..23}
// Lane l>=16: row = base_row + l-16, K-chunk {8..15, 24..31}
__device__ __forceinline__ v16bf load_frag(const __bf16* __restrict__ base,
                                           int row, int ld, int k0, int lane) {
  int r  = row + (lane & 15);
  int ks = (lane >> 4) << 3;
  const __bf16* p = base + r * ld + k0 + ks;
  v8bf lo = *(const v8bf*)p;          // K = k0+ks    .. +7
  v8bf hi = *(const v8bf*)(p + 16);   // K = k0+ks+16 .. +23
  return __builtin_shufflevector(lo, hi, 0,1,2,3,4,5,6,7,8,9,10,11,12,13,14,15);
}

// ---------- fp32 -> bf16 conversion ----------

__global__ void __launch_bounds__(256)
cvt_f32_bf16(const float* __restrict__ src, __bf16* __restrict__ dst, int n) {
  int i = blockIdx.x * blockDim.x + threadIdx.x;
  int stride = gridDim.x * blockDim.x;
  for (; i < n; i += stride) dst[i] = to_bf16(src[i]);
}

// ---------- projection GEMM: C[M=4096,N=1024] = A[M,K] * W[N,K]^T + bias ----------
// MODE 0: bf16 out, layout [B,H,T,D]  (Q with scale, K with scale=1)
// MODE 1: bf16 out, layout [B,H,D,T]  (V transposed for P*V WMMA)
// MODE 2: fp32 out, row-major [M,N]   (final output projection)

template <int MODE>
__global__ void __launch_bounds__(256)
gemm_proj(const __bf16* __restrict__ A, const __bf16* __restrict__ W,
          const float* __restrict__ bias, float scale,
          __bf16* __restrict__ outB, float* __restrict__ outF) {
  const int K = EMB, N = EMB;
  int lane = threadIdx.x & 31;
  int wave = threadIdx.x >> 5;
  int wm = wave >> 1, wn = wave & 1;              // 4x2 wave grid
  int m0 = blockIdx.x * 128 + wm * 32;            // 32 rows per wave
  int n0 = blockIdx.y * 128 + wn * 64;            // 64 cols per wave

  v8f acc[2][4] = {};
  for (int k0 = 0; k0 < K; k0 += 32) {
    v16bf a0 = load_frag(A, m0,      K, k0, lane);
    v16bf a1 = load_frag(A, m0 + 16, K, k0, lane);
    v16bf b[4];
#pragma unroll
    for (int ni = 0; ni < 4; ++ni) b[ni] = load_frag(W, n0 + 16 * ni, K, k0, lane);
#pragma unroll
    for (int ni = 0; ni < 4; ++ni) {
      acc[0][ni] = __builtin_amdgcn_wmma_f32_16x16x32_bf16(false, a0, false, b[ni],
                                                           (short)0, acc[0][ni], false, false);
      acc[1][ni] = __builtin_amdgcn_wmma_f32_16x16x32_bf16(false, a1, false, b[ni],
                                                           (short)0, acc[1][ni], false, false);
    }
  }

  // Epilogue: lane holds column n = n_tile + (lane&15), rows m = m_tile + (lane>>4)*8 + vg
  int hi = lane >> 4, col = lane & 15;
#pragma unroll
  for (int ni = 0; ni < 4; ++ni) {
    int n = n0 + ni * 16 + col;
    float bn = bias[n];
    if constexpr (MODE == 2) {
#pragma unroll
      for (int mi = 0; mi < 2; ++mi) {
        float* op = outF + (m0 + mi * 16 + hi * 8) * N + n;
#pragma unroll
        for (int vg = 0; vg < 8; ++vg) op[vg * N] = acc[mi][ni][vg] + bn;
      }
    } else {
      int h = n >> 6, d = n & 63;
#pragma unroll
      for (int mi = 0; mi < 2; ++mi) {
        int mb = m0 + mi * 16 + hi * 8;
#pragma unroll
        for (int vg = 0; vg < 8; ++vg) {
          int m = mb + vg;
          int t = m >> 1, bb = m & 1;             // x rows are (t*B + b)
          float c = (acc[mi][ni][vg] + bn) * scale;
          if constexpr (MODE == 0)
            outB[((((bb * HEADS + h) * T_LEN) + t) << 6) + d] = to_bf16(c);
          else
            outB[(((bb * HEADS + h) * HDIM + d) * T_LEN) + t] = to_bf16(c);
        }
      }
    }
  }
}

// ---------- flash attention: per wave 16 query rows, transposed-scores trick ----------
// Sᵀ = K·Qᵀ lands in the exact B-operand layout needed by Oᵀ += Vᵀ·Pᵀ, so the
// softmaxed probabilities feed the second WMMA with zero LDS/shuffle transposes.

__global__ void __launch_bounds__(256)
flash_attn(const __bf16* __restrict__ Qb, const __bf16* __restrict__ Kb,
           const __bf16* __restrict__ Vtb, __bf16* __restrict__ Ab) {
  const float LG2E = 1.44269504089f;
  int lane = threadIdx.x & 31;
  int wave = threadIdx.x >> 5;
  int t0 = blockIdx.x * 128 + wave * 16;          // 16 query rows per wave
  int bh = blockIdx.y;                            // b*HEADS + h
  const __bf16* Qh  = Qb  + (size_t)bh * T_LEN * HDIM;
  const __bf16* Kh  = Kb  + (size_t)bh * T_LEN * HDIM;
  const __bf16* Vth = Vtb + (size_t)bh * HDIM * T_LEN;

  // Qᵀ B-operand fragments (lane = t column, K = d), loaded once
  v16bf bq[2];
#pragma unroll
  for (int kk = 0; kk < 2; ++kk) bq[kk] = load_frag(Qh, t0, HDIM, kk * 32, lane);

  v8f o[4] = {};            // Oᵀ[d=64, t=16] : 4 M-tiles over d
  float m = -3.0e38f, l = 0.0f;

  for (int s0 = 0; s0 < T_LEN; s0 += 32) {
    // Sᵀ = K·Qᵀ : two 16-wide s tiles, K-dim = 64 (2 WMMA k-steps)
    v8f st0 = {}, st1 = {};
#pragma unroll
    for (int kk = 0; kk < 2; ++kk) {
      v16bf ak0 = load_frag(Kh, s0,      HDIM, kk * 32, lane);
      v16bf ak1 = load_frag(Kh, s0 + 16, HDIM, kk * 32, lane);
      st0 = __builtin_amdgcn_wmma_f32_16x16x32_bf16(false, ak0, false, bq[kk],
                                                    (short)0, st0, false, false);
      st1 = __builtin_amdgcn_wmma_f32_16x16x32_bf16(false, ak1, false, bq[kk],
                                                    (short)0, st1, false, false);
    }
    __builtin_prefetch(Kh + (s0 + 32) * HDIM, 0, 1);
    __builtin_prefetch(Vth + s0 + 32, 0, 1);

    // online softmax over s for this lane's t column; lane pair (j, j^16)
    // together covers all 32 s values of the same t -> single xor-16 shuffle
    float mloc = -3.0e38f;
#pragma unroll
    for (int i = 0; i < 8; ++i) mloc = fmaxf(mloc, fmaxf(st0[i], st1[i]));
    mloc = fmaxf(mloc, __shfl_xor(mloc, 16, 32));
    float mnew = fmaxf(m, mloc);
    float alpha = __builtin_amdgcn_exp2f((m - mnew) * LG2E);
    m = mnew;

    v8bf p0, p1;
    float ssum = 0.0f;
#pragma unroll
    for (int i = 0; i < 8; ++i) {
      float e0 = __builtin_amdgcn_exp2f((st0[i] - mnew) * LG2E);
      float e1 = __builtin_amdgcn_exp2f((st1[i] - mnew) * LG2E);
      p0[i] = to_bf16(e0);
      p1[i] = to_bf16(e1);
      ssum += e0 + e1;
    }
    ssum += __shfl_xor(ssum, 16, 32);
    l = l * alpha + ssum;

#pragma unroll
    for (int dt = 0; dt < 4; ++dt)
#pragma unroll
      for (int i = 0; i < 8; ++i) o[dt][i] *= alpha;

    // Pᵀ already in B-operand lane/K layout
    v16bf pb = __builtin_shufflevector(p0, p1, 0,1,2,3,4,5,6,7,8,9,10,11,12,13,14,15);
#pragma unroll
    for (int dt = 0; dt < 4; ++dt) {
      v16bf av = load_frag(Vth, dt * 16, T_LEN, s0, lane);   // A: rows d, K = s
      o[dt] = __builtin_amdgcn_wmma_f32_16x16x32_bf16(false, av, false, pb,
                                                      (short)0, o[dt], false, false);
    }
  }

  // normalize and store attn (bf16) into [T, B, E]; d runs over vg -> contiguous b128
  float inv = 1.0f / l;
  int col = lane & 15, hi = lane >> 4;
  int t = t0 + col;
  int b = bh >> 4, h = bh & 15;
  __bf16* outrow = Ab + ((size_t)t * BATCH + b) * EMB + h * HDIM + hi * 8;
#pragma unroll
  for (int dt = 0; dt < 4; ++dt) {
    v8bf ov;
#pragma unroll
    for (int i = 0; i < 8; ++i) ov[i] = to_bf16(o[dt][i] * inv);
    *(v8bf*)(outrow + dt * 16) = ov;
  }
}

// ---------- host launch ----------

extern "C" void kernel_launch(void* const* d_in, const int* in_sizes, int n_in,
                              void* d_out, int out_size, void* d_ws, size_t ws_size,
                              hipStream_t stream) {
  const float* x  = (const float*)d_in[0];
  const float* wq = (const float*)d_in[1];
  const float* bq = (const float*)d_in[2];
  const float* wk = (const float*)d_in[3];
  const float* bk = (const float*)d_in[4];
  const float* wv = (const float*)d_in[5];
  const float* bv = (const float*)d_in[6];
  const float* wo = (const float*)d_in[7];
  const float* bo = (const float*)d_in[8];
  (void)in_sizes; (void)n_in; (void)out_size; (void)ws_size;

  char* ws = (char*)d_ws;
  const size_t MB = (size_t)1 << 20;
  __bf16* xb  = (__bf16*)(ws + 0 * MB);    // [4096,1024] bf16 (8 MiB)
  __bf16* wqb = (__bf16*)(ws + 8 * MB);    // 2 MiB each
  __bf16* wkb = (__bf16*)(ws + 10 * MB);
  __bf16* wvb = (__bf16*)(ws + 12 * MB);
  __bf16* wob = (__bf16*)(ws + 14 * MB);
  __bf16* Qb  = (__bf16*)(ws + 16 * MB);   // [B,H,T,D]  8 MiB
  __bf16* Kb  = (__bf16*)(ws + 24 * MB);   // [B,H,T,D]  8 MiB
  __bf16* Vtb = (__bf16*)(ws + 32 * MB);   // [B,H,D,T]  8 MiB
  __bf16* Ab  = (__bf16*)(ws + 40 * MB);   // [T,B,E]    8 MiB

  cvt_f32_bf16<<<2048, 256, 0, stream>>>(x,  xb,  T_LEN * BATCH * EMB);
  cvt_f32_bf16<<<1024, 256, 0, stream>>>(wq, wqb, EMB * EMB);
  cvt_f32_bf16<<<1024, 256, 0, stream>>>(wk, wkb, EMB * EMB);
  cvt_f32_bf16<<<1024, 256, 0, stream>>>(wv, wvb, EMB * EMB);
  cvt_f32_bf16<<<1024, 256, 0, stream>>>(wo, wob, EMB * EMB);

  dim3 g(32, 8);   // M=4096 / 128, N=1024 / 128
  gemm_proj<0><<<g, 256, 0, stream>>>(xb, wqb, bq, 0.125f, Qb,  nullptr);  // Q * D^-0.5
  gemm_proj<0><<<g, 256, 0, stream>>>(xb, wkb, bk, 1.0f,   Kb,  nullptr);
  gemm_proj<1><<<g, 256, 0, stream>>>(xb, wvb, bv, 1.0f,   Vtb, nullptr);  // V transposed

  flash_attn<<<dim3(16, 32), 256, 0, stream>>>(Qb, Kb, Vtb, Ab);

  gemm_proj<2><<<g, 256, 0, stream>>>(Ab, wob, bo, 1.0f, nullptr, (float*)d_out);
}
